// HeterogeneousGAT_9818295239156
// MI455X (gfx1250) — matrix operand order repaired and checked by reading, not compiled, not run on hardware
//
#include <hip/hip_runtime.h>
#include <hip/hip_bf16.h>
#include <math.h>

#define TDH 128   // feature dim
#define THD 4     // heads

typedef __attribute__((ext_vector_type(16))) __bf16 v16bf;
typedef __attribute__((ext_vector_type(8)))  float  v8f;

// ---------- small device helpers ----------
__device__ __forceinline__ unsigned short f2bf(float f) {
  unsigned u = __float_as_uint(f);
  u += 0x7FFFu + ((u >> 16) & 1u);     // round-to-nearest-even
  return (unsigned short)(u >> 16);
}
// monotone float<->uint mapping so integer atomicMax == float max
__device__ __forceinline__ unsigned fenc(float f) {
  unsigned u = __float_as_uint(f);
  return (u & 0x80000000u) ? ~u : (u | 0x80000000u);
}
__device__ __forceinline__ float fdec(unsigned u) {
  return (u & 0x80000000u) ? __uint_as_float(u ^ 0x80000000u) : __uint_as_float(~u);
}
__device__ __forceinline__ float lrelu(float v) { return v > 0.f ? v : 0.2f * v; }

// ---------- fills ----------
__global__ void k_fill_f(float* p, float v, long n) {
  long i = (long)blockIdx.x * 256 + threadIdx.x;
  if (i < n) p[i] = v;
}
__global__ void k_fill_u(unsigned* p, unsigned v, long n) {
  long i = (long)blockIdx.x * 256 + threadIdx.x;
  if (i < n) p[i] = v;
}

// ---------- weight prep: W[k][n] f32 -> Wt[n][k] bf16 ----------
__global__ void k_prep_w(const float* __restrict__ W, unsigned short* __restrict__ Wt) {
  int idx = blockIdx.x * 256 + threadIdx.x;
  if (idx < TDH * TDH) {
    int n = idx >> 7, k = idx & 127;
    Wt[idx] = f2bf(W[k * TDH + n]);
  }
}

// ---------- WMMA GEMM (+fused attention scores) ----------
// Out[M,128] = A[M,128] @ W (+bias); Wt = W^T bf16.
// If attS != nullptr also emits aoutS[n][h] = dot(Out_nobias[n][h*32:+32], attS[h])
// and likewise aoutD from attD (reference computes scores pre-bias; att path is
// only used for the bias-less layer GEMMs).
__global__ __launch_bounds__(256) void k_gemm_wmma(
    const float* __restrict__ A, const unsigned short* __restrict__ Wt,
    const float* __restrict__ bias, float* __restrict__ Out,
    const float* __restrict__ attS, const float* __restrict__ attD,
    float* __restrict__ aoutS, float* __restrict__ aoutD, int M) {
  __shared__ __align__(16) unsigned short sA[32][TDH];   // 8 KB
  __shared__ __align__(16) unsigned short sW[TDH][TDH];  // 32 KB
  __shared__ __align__(16) float          sO[32][TDH];   // 16 KB
  const int tid  = threadIdx.x;
  const int row0 = blockIdx.x * 32;

  // stage A stripe (32x128 f32 -> bf16)
  for (int i = tid; i < 32 * TDH / 4; i += 256) {
    int r = i >> 5, c4 = (i & 31) * 4;
    float4 v = make_float4(0.f, 0.f, 0.f, 0.f);
    if (row0 + r < M) v = *(const float4*)(A + (size_t)(row0 + r) * TDH + c4);
    sA[r][c4 + 0] = f2bf(v.x); sA[r][c4 + 1] = f2bf(v.y);
    sA[r][c4 + 2] = f2bf(v.z); sA[r][c4 + 3] = f2bf(v.w);
  }
  // stage Wt (128x128 bf16 = 32KB)
  {
    const uint4* g = (const uint4*)Wt;
    uint4* s = (uint4*)&sW[0][0];
    for (int i = tid; i < TDH * TDH / 8; i += 256) s[i] = g[i];
  }
  __syncthreads();

  const int lane = tid & 31, wv = tid >> 5;
  const int m  = lane & 15;
  const int hi = lane >> 4;           // 0: lanes 0-15, 1: lanes 16-31
  const int n  = (wv << 4) + m;       // output column

  union { v16bf v; unsigned u[8]; } a0, a1, b;
  v8f c0, c1;
  #pragma unroll
  for (int i = 0; i < 8; i++) { c0[i] = 0.f; c1[i] = 0.f; }

  #pragma unroll
  for (int ks = 0; ks < 4; ks++) {
    const int k0 = ks * 32;
    #pragma unroll
    for (int e = 0; e < 8; e++) {
      // A 16x32 bf16 layout: lane<16 K in {0..7,16..23}, lane>=16 +8; pairs packed/VGPR
      int kk = ((e < 4) ? 2 * e : 8 + 2 * e) + 8 * hi;
      a0.u[e] = *(const unsigned*)&sA[m][k0 + kk];
      a1.u[e] = *(const unsigned*)&sA[16 + m][k0 + kk];
      // B 32x16 bf16 layout: same K order as A, for column n
      b.u[e]  = *(const unsigned*)&sW[n][k0 + kk];
    }
    c0 = __builtin_amdgcn_wmma_f32_16x16x32_bf16(false, a0.v, false, b.v, (short)0, c0,
                                                 false, false);
    c1 = __builtin_amdgcn_wmma_f32_16x16x32_bf16(false, a1.v, false, b.v, (short)0, c1,
                                                 false, false);
  }
  const float bb = bias ? bias[n] : 0.f;
  #pragma unroll
  for (int r = 0; r < 8; r++) {
    // D: VGPR r -> M=r (lanes<16) / r+8 (lanes>=16)
    int lr0 = r + 8 * hi, lr1 = 16 + r + 8 * hi;
    sO[lr0][n] = c0[r] + bb;
    sO[lr1][n] = c1[r] + bb;
    if (row0 + lr0 < M) Out[(size_t)(row0 + lr0) * TDH + n] = c0[r] + bb;
    if (row0 + lr1 < M) Out[(size_t)(row0 + lr1) * TDH + n] = c1[r] + bb;
  }

  // fused per-head attention scores from the f32 tile still resident in LDS
  if (attS) {
    __syncthreads();
    int half = tid >> 7;            // 0 -> src scores, 1 -> dst scores
    int item = tid & 127;           // 32 rows x 4 heads
    int rr = item >> 2, h = item & 3;
    const float* av = (half ? attD : attS) + h * 32;
    const float* o  = &sO[rr][h * 32];
    float s = 0.f;
    #pragma unroll
    for (int i = 0; i < 32; i++) s += o[i] * av[i];
    int grow = row0 + rr;
    if (grow < M) (half ? aoutD : aoutS)[grow * THD + h] = s;
  }
}

// ---------- segment-softmax passes (one thread per edge, float4 over heads) ----------
__global__ void k_emax(const int* __restrict__ ei, int E, const float* __restrict__ asrc,
                       const float* __restrict__ adst, unsigned* __restrict__ mb) {
  int e = blockIdx.x * 256 + threadIdx.x;
  if (e >= E) return;
  int s = ei[e], d = ei[E + e];
  float4 a = *(const float4*)(asrc + (size_t)s * THD);
  float4 b = *(const float4*)(adst + (size_t)d * THD);
  unsigned* mp = mb + (size_t)d * THD;
  atomicMax(mp + 0, fenc(lrelu(a.x + b.x)));
  atomicMax(mp + 1, fenc(lrelu(a.y + b.y)));
  atomicMax(mp + 2, fenc(lrelu(a.z + b.z)));
  atomicMax(mp + 3, fenc(lrelu(a.w + b.w)));
}
__global__ void k_smax(const float* __restrict__ asrc, const float* __restrict__ adst,
                       unsigned* __restrict__ mb, int N) {
  int n = blockIdx.x * 256 + threadIdx.x;
  if (n >= N) return;
  float4 a = *(const float4*)(asrc + (size_t)n * THD);
  float4 b = *(const float4*)(adst + (size_t)n * THD);
  unsigned* mp = mb + (size_t)n * THD;
  atomicMax(mp + 0, fenc(lrelu(a.x + b.x)));
  atomicMax(mp + 1, fenc(lrelu(a.y + b.y)));
  atomicMax(mp + 2, fenc(lrelu(a.z + b.z)));
  atomicMax(mp + 3, fenc(lrelu(a.w + b.w)));
}
__global__ void k_esum(const int* __restrict__ ei, int E, const float* __restrict__ asrc,
                       const float* __restrict__ adst, const unsigned* __restrict__ mb,
                       float* __restrict__ zb) {
  int e = blockIdx.x * 256 + threadIdx.x;
  if (e >= E) return;
  int s = ei[e], d = ei[E + e];
  float4 a = *(const float4*)(asrc + (size_t)s * THD);
  float4 b = *(const float4*)(adst + (size_t)d * THD);
  const unsigned* mp = mb + (size_t)d * THD;
  float* zp = zb + (size_t)d * THD;
  atomicAdd(zp + 0, __expf(lrelu(a.x + b.x) - fdec(mp[0])));
  atomicAdd(zp + 1, __expf(lrelu(a.y + b.y) - fdec(mp[1])));
  atomicAdd(zp + 2, __expf(lrelu(a.z + b.z) - fdec(mp[2])));
  atomicAdd(zp + 3, __expf(lrelu(a.w + b.w) - fdec(mp[3])));
}
__global__ void k_ssum(const float* __restrict__ asrc, const float* __restrict__ adst,
                       const unsigned* __restrict__ mb, float* __restrict__ zb, int N) {
  int n = blockIdx.x * 256 + threadIdx.x;
  if (n >= N) return;
  float4 a = *(const float4*)(asrc + (size_t)n * THD);
  float4 b = *(const float4*)(adst + (size_t)n * THD);
  const unsigned* mp = mb + (size_t)n * THD;
  float* zp = zb + (size_t)n * THD;
  atomicAdd(zp + 0, __expf(lrelu(a.x + b.x) - fdec(mp[0])));
  atomicAdd(zp + 1, __expf(lrelu(a.y + b.y) - fdec(mp[1])));
  atomicAdd(zp + 2, __expf(lrelu(a.z + b.z) - fdec(mp[2])));
  atomicAdd(zp + 3, __expf(lrelu(a.w + b.w) - fdec(mp[3])));
}
// one wave per edge: acc[dst] += alpha_h * feat[src]
__global__ __launch_bounds__(256) void k_escat(
    const int* __restrict__ ei, int E, const float* __restrict__ asrc,
    const float* __restrict__ adst, const unsigned* __restrict__ mb,
    const float* __restrict__ zb, const float* __restrict__ feat,
    float* __restrict__ acc) {
  int e = blockIdx.x * 8 + (threadIdx.x >> 5);
  if (e >= E) return;
  int lane = threadIdx.x & 31;
  int s = ei[e], d = ei[E + e];
  int c = lane * 4, h = c >> 5;
  float v = lrelu(asrc[s * THD + h] + adst[d * THD + h]);
  float alpha = __expf(v - fdec(mb[d * THD + h])) / (zb[d * THD + h] + 1e-16f);
  float4 f = *(const float4*)(feat + (size_t)s * TDH + c);
  float* o = acc + (size_t)d * TDH + c;
  atomicAdd(o + 0, alpha * f.x); atomicAdd(o + 1, alpha * f.y);
  atomicAdd(o + 2, alpha * f.z); atomicAdd(o + 3, alpha * f.w);
}
__global__ __launch_bounds__(256) void k_sscat(
    const float* __restrict__ asrc, const float* __restrict__ adst,
    const unsigned* __restrict__ mb, const float* __restrict__ zb,
    const float* __restrict__ feat, float* __restrict__ acc, int N) {
  int n = blockIdx.x * 8 + (threadIdx.x >> 5);
  if (n >= N) return;
  int lane = threadIdx.x & 31;
  int c = lane * 4, h = c >> 5;
  float v = lrelu(asrc[n * THD + h] + adst[n * THD + h]);
  float alpha = __expf(v - fdec(mb[n * THD + h])) / (zb[n * THD + h] + 1e-16f);
  float4 f = *(const float4*)(feat + (size_t)n * TDH + c);
  float* o = acc + (size_t)n * TDH + c;
  atomicAdd(o + 0, alpha * f.x); atomicAdd(o + 1, alpha * f.y);
  atomicAdd(o + 2, alpha * f.z); atomicAdd(o + 3, alpha * f.w);
}

// ---------- layer-1 fusion: relation attention + LN + ELU ----------
__global__ __launch_bounds__(128) void k_fuse1(
    const float* __restrict__ acc, const float* __restrict__ b1,
    const float* __restrict__ Wr, const float* __restrict__ br,
    const float* __restrict__ g, const float* __restrict__ be,
    float* __restrict__ H, int N) {
  __shared__ float red[128];
  __shared__ float sc[4];
  const int n = blockIdx.x, c = threadIdx.x;
  const size_t nf = (size_t)N * TDH;
  float o0 = acc[(size_t)n * TDH + c] + b1[c];
  float o1 = acc[nf + (size_t)n * TDH + c] + b1[TDH + c];
  float wr = Wr[c];
  red[c] = o0 * wr; __syncthreads();
  for (int s = 64; s > 0; s >>= 1) { if (c < s) red[c] += red[c + s]; __syncthreads(); }
  if (c == 0) sc[0] = red[0] + br[0];
  __syncthreads();
  red[c] = o1 * wr; __syncthreads();
  for (int s = 64; s > 0; s >>= 1) { if (c < s) red[c] += red[c + s]; __syncthreads(); }
  if (c == 0) sc[1] = red[0] + br[0];
  __syncthreads();
  float l0 = sc[0], l1 = sc[1];
  float mx = fmaxf(l0, l1);
  float e0 = __expf(l0 - mx), e1 = __expf(l1 - mx);
  float inv = 1.f / (e0 + e1);
  float h = (e0 * inv) * o0 + (e1 * inv) * o1;
  red[c] = h; __syncthreads();
  for (int s = 64; s > 0; s >>= 1) { if (c < s) red[c] += red[c + s]; __syncthreads(); }
  if (c == 0) sc[2] = red[0] * (1.f / TDH);
  __syncthreads();
  float mu = sc[2], d = h - mu;
  red[c] = d * d; __syncthreads();
  for (int s = 64; s > 0; s >>= 1) { if (c < s) red[c] += red[c + s]; __syncthreads(); }
  if (c == 0) sc[3] = red[0] * (1.f / TDH);
  __syncthreads();
  float y = d * rsqrtf(sc[3] + 1e-5f) * g[c] + be[c];
  y = y > 0.f ? y : (__expf(y) - 1.f);   // ELU
  H[(size_t)n * TDH + c] = y;
}

// ---------- layer-2 fusion: residual + mean of relations + LN + ELU ----------
__global__ __launch_bounds__(128) void k_fuse2(
    float* __restrict__ H, const float* __restrict__ acc, const float* __restrict__ b2,
    const float* __restrict__ g, const float* __restrict__ be, int N) {
  __shared__ float red[128];
  __shared__ float sc[2];
  const int n = blockIdx.x, c = threadIdx.x;
  const size_t nf = (size_t)N * TDH;
  float h2 = (acc[(size_t)n * TDH + c] + b2[c] +
              acc[nf + (size_t)n * TDH + c] + b2[TDH + c]) * 0.5f;
  float h = H[(size_t)n * TDH + c] + h2;
  red[c] = h; __syncthreads();
  for (int s = 64; s > 0; s >>= 1) { if (c < s) red[c] += red[c + s]; __syncthreads(); }
  if (c == 0) sc[0] = red[0] * (1.f / TDH);
  __syncthreads();
  float mu = sc[0], d = h - mu;
  red[c] = d * d; __syncthreads();
  for (int s = 64; s > 0; s >>= 1) { if (c < s) red[c] += red[c + s]; __syncthreads(); }
  if (c == 0) sc[1] = red[0] * (1.f / TDH);
  __syncthreads();
  float y = d * rsqrtf(sc[1] + 1e-5f) * g[c] + be[c];
  y = y > 0.f ? y : (__expf(y) - 1.f);
  H[(size_t)n * TDH + c] = y;
}

// ---------- final LN + L2 normalize + global pool accumulate ----------
__global__ __launch_bounds__(128) void k_finish(
    const float* __restrict__ T, const float* __restrict__ g, const float* __restrict__ be,
    float* __restrict__ out, float* __restrict__ gsum, unsigned* __restrict__ gmax, int N) {
  __shared__ float red[128];
  __shared__ float sc[3];
  const int n = blockIdx.x, c = threadIdx.x;
  float t = T[(size_t)n * TDH + c];
  red[c] = t; __syncthreads();
  for (int s = 64; s > 0; s >>= 1) { if (c < s) red[c] += red[c + s]; __syncthreads(); }
  if (c == 0) sc[0] = red[0] * (1.f / TDH);
  __syncthreads();
  float mu = sc[0], d = t - mu;
  red[c] = d * d; __syncthreads();
  for (int s = 64; s > 0; s >>= 1) { if (c < s) red[c] += red[c + s]; __syncthreads(); }
  if (c == 0) sc[1] = red[0] * (1.f / TDH);
  __syncthreads();
  float y = d * rsqrtf(sc[1] + 1e-5f) * g[c] + be[c];
  red[c] = y * y; __syncthreads();
  for (int s = 64; s > 0; s >>= 1) { if (c < s) red[c] += red[c + s]; __syncthreads(); }
  if (c == 0) sc[2] = sqrtf(red[0]);
  __syncthreads();
  y = y / fmaxf(sc[2], 1e-12f);
  out[(size_t)n * TDH + c] = y;
  atomicAdd(&gsum[c], y);
  atomicMax(&gmax[c], fenc(y));
}
__global__ void k_glob(float* __restrict__ out, const float* __restrict__ gsum,
                       const unsigned* __restrict__ gmax, int N) {
  int c = threadIdx.x;
  if (c < 128)       out[(size_t)N * TDH + c] = gsum[c] / (float)N;
  else if (c < 256)  out[(size_t)N * TDH + c] = fdec(gmax[c - 128]);
}

// ============================ host launcher ============================
extern "C" void kernel_launch(void* const* d_in, const int* in_sizes, int n_in,
                              void* d_out, int out_size, void* d_ws, size_t ws_size,
                              hipStream_t stream) {
  const float* x   = (const float*)d_in[0];
  const int*   ei0 = (const int*)  d_in[1];
  const int*   ei1 = (const int*)  d_in[2];
  const float* W1  = (const float*)d_in[5];
  const float* as1 = (const float*)d_in[6];
  const float* ad1 = (const float*)d_in[7];
  const float* b1  = (const float*)d_in[8];
  const float* Wr  = (const float*)d_in[9];
  const float* br  = (const float*)d_in[10];
  const float* W2  = (const float*)d_in[11];
  const float* as2 = (const float*)d_in[12];
  const float* ad2 = (const float*)d_in[13];
  const float* b2  = (const float*)d_in[14];
  const float* W3  = (const float*)d_in[15];
  const float* b3  = (const float*)d_in[16];
  const float* g1  = (const float*)d_in[17];
  const float* be1 = (const float*)d_in[18];
  const float* g2  = (const float*)d_in[19];
  const float* be2 = (const float*)d_in[20];
  const float* g3  = (const float*)d_in[21];
  const float* be3 = (const float*)d_in[22];

  const int N  = in_sizes[0] / TDH;
  const int E0 = in_sizes[1] / 2;
  const int E1 = in_sizes[2] / 2;
  const size_t nf  = (size_t)N * TDH;
  const size_t na  = (size_t)N * THD;
  float* out = (float*)d_out;

  // workspace layout (floats)
  float*    FEAT = (float*)d_ws;            // 2*nf transformed features per relation
  float*    ACC  = FEAT + 2 * nf;           // 2*nf scatter accumulators
  float*    H    = ACC  + 2 * nf;           // nf   node state
  float*    ASRC = H    + nf;               // 2*na
  float*    ADST = ASRC + 2 * na;           // 2*na
  unsigned* MB   = (unsigned*)(ADST + 2 * na); // 2*na softmax max (encoded)
  float*    ZB   = (float*)(MB + 2 * na);   // 2*na softmax denom
  float*    GSUM = ZB + 2 * na;             // 128
  unsigned* GMAX = (unsigned*)(GSUM + 128); // 128
  unsigned short* WT = (unsigned short*)(GMAX + 128); // 5 * 16384 bf16 transposed weights

  const unsigned NEGINF_ENC = 0x007FFFFFu;  // fenc(-inf)
  auto cdiv = [](long a, long b) { return (int)((a + b - 1) / b); };
  const int gmm = cdiv(N, 32);

  // weight prep (transpose + bf16)
  k_prep_w<<<64, 256, 0, stream>>>(W1,          WT);
  k_prep_w<<<64, 256, 0, stream>>>(W1 + 16384,  WT + 16384);
  k_prep_w<<<64, 256, 0, stream>>>(W2,          WT + 2 * 16384);
  k_prep_w<<<64, 256, 0, stream>>>(W2 + 16384,  WT + 3 * 16384);
  k_prep_w<<<64, 256, 0, stream>>>(W3,          WT + 4 * 16384);

  const int* eis[2] = { ei0, ei1 };
  const int  Es [2] = { E0, E1 };

  // ---------------- layer 1 ----------------
  for (int r = 0; r < 2; r++)
    k_gemm_wmma<<<gmm, 256, 0, stream>>>(x, WT + r * 16384, nullptr, FEAT + r * nf,
                                         as1 + r * TDH, ad1 + r * TDH,
                                         ASRC + r * na, ADST + r * na, N);
  k_fill_f<<<cdiv(2 * nf, 256), 256, 0, stream>>>(ACC, 0.f, 2 * nf);
  k_fill_f<<<cdiv(2 * na, 256), 256, 0, stream>>>(ZB, 0.f, 2 * na);
  k_fill_u<<<cdiv(2 * na, 256), 256, 0, stream>>>(MB, NEGINF_ENC, 2 * na);
  for (int r = 0; r < 2; r++) {
    k_emax <<<cdiv(Es[r], 256), 256, 0, stream>>>(eis[r], Es[r], ASRC + r * na, ADST + r * na, MB + r * na);
    k_esum <<<cdiv(Es[r], 256), 256, 0, stream>>>(eis[r], Es[r], ASRC + r * na, ADST + r * na, MB + r * na, ZB + r * na);
    k_escat<<<cdiv(Es[r], 8), 256, 0, stream>>>(eis[r], Es[r], ASRC + r * na, ADST + r * na, MB + r * na, ZB + r * na, FEAT + r * nf, ACC + r * nf);
  }
  k_fuse1<<<N, 128, 0, stream>>>(ACC, b1, Wr, br, g1, be1, H, N);

  // ---------------- layer 2 (self loops, mean over relations, residual) ----------------
  for (int r = 0; r < 2; r++)
    k_gemm_wmma<<<gmm, 256, 0, stream>>>(H, WT + (2 + r) * 16384, nullptr, FEAT + r * nf,
                                         as2 + r * TDH, ad2 + r * TDH,
                                         ASRC + r * na, ADST + r * na, N);
  k_fill_f<<<cdiv(2 * nf, 256), 256, 0, stream>>>(ACC, 0.f, 2 * nf);
  k_fill_f<<<cdiv(2 * na, 256), 256, 0, stream>>>(ZB, 0.f, 2 * na);
  k_fill_u<<<cdiv(2 * na, 256), 256, 0, stream>>>(MB, NEGINF_ENC, 2 * na);
  for (int r = 0; r < 2; r++) {
    k_emax <<<cdiv(Es[r], 256), 256, 0, stream>>>(eis[r], Es[r], ASRC + r * na, ADST + r * na, MB + r * na);
    k_smax <<<cdiv(N, 256), 256, 0, stream>>>(ASRC + r * na, ADST + r * na, MB + r * na, N);
    k_esum <<<cdiv(Es[r], 256), 256, 0, stream>>>(eis[r], Es[r], ASRC + r * na, ADST + r * na, MB + r * na, ZB + r * na);
    k_ssum <<<cdiv(N, 256), 256, 0, stream>>>(ASRC + r * na, ADST + r * na, MB + r * na, ZB + r * na, N);
    k_escat<<<cdiv(Es[r], 8), 256, 0, stream>>>(eis[r], Es[r], ASRC + r * na, ADST + r * na, MB + r * na, ZB + r * na, FEAT + r * nf, ACC + r * nf);
    k_sscat<<<cdiv(N, 8), 256, 0, stream>>>(ASRC + r * na, ADST + r * na, MB + r * na, ZB + r * na, FEAT + r * nf, ACC + r * nf, N);
  }
  k_fuse2<<<N, 128, 0, stream>>>(H, ACC, b2, g2, be2, N);

  // ---------------- head: h@W3+b3, LN, L2 normalize, global mean/max ----------------
  k_gemm_wmma<<<gmm, 256, 0, stream>>>(H, WT + 4 * 16384, b3, FEAT,
                                       nullptr, nullptr, nullptr, nullptr, N);
  k_fill_f<<<1, 256, 0, stream>>>(GSUM, 0.f, 128);
  k_fill_u<<<1, 256, 0, stream>>>(GMAX, NEGINF_ENC, 128);
  k_finish<<<N, 128, 0, stream>>>(FEAT, g3, be3, out, GSUM, GMAX, N);
  k_glob<<<1, 256, 0, stream>>>(out, GSUM, GMAX, N);
}
